// PyTorchESN_68487548502042
// MI455X (gfx1250) — matrix operand order repaired and checked by reading, not compiled, not run on hardware
//
#include <hip/hip_runtime.h>
#include <cstdint>
#include <cstddef>

// Problem constants (fixed by setup_inputs)
#define B_    256
#define T_    512
#define D_    64
#define R_    2048
#define O_    10
#define K_    2112          // D_ + R_ combined GEMM K
#define LRATE 0.5f

typedef __attribute__((ext_vector_type(16))) __bf16       v16bf;
typedef __attribute__((ext_vector_type(8)))  float        v8f;
typedef __attribute__((ext_vector_type(4)))  unsigned int u32x4;
typedef __attribute__((ext_vector_type(4)))  int          i32x4;

union FragAB { v16bf v; u32x4 q[2]; };

// CDNA5 async global->LDS path (gfx1250); guarded so the host pass and older
// toolchains parse the sync fallback instead.
#if defined(__HIP_DEVICE_COMPILE__) && __has_builtin(__builtin_amdgcn_global_load_async_to_lds_b128)
#define USE_ASYNC_COPY 1
#else
#define USE_ASYNC_COPY 0
#endif

#if USE_ASYNC_COPY
typedef __attribute__((address_space(1))) i32x4* gvec4p;   // global v4i*
typedef __attribute__((address_space(3))) i32x4* lvec4p;   // LDS v4i*
#endif

__device__ __forceinline__ void copy16_g2l(__bf16* lds_dst, const __bf16* gsrc) {
#if USE_ASYNC_COPY
  __builtin_amdgcn_global_load_async_to_lds_b128((gvec4p)gsrc, (lvec4p)lds_dst, 0, 0);
#else
  *(u32x4*)lds_dst = *(const u32x4*)gsrc;
#endif
}

__device__ __forceinline__ void wait_async_copies() {
#if USE_ASYNC_COPY
# if __has_builtin(__builtin_amdgcn_s_wait_asynccnt)
  __builtin_amdgcn_s_wait_asynccnt(0);
# else
  asm volatile("s_wait_asynccnt 0" ::: "memory");
# endif
#endif
}

// ---------------- one-time prep kernels ----------------

// Wcat[n][k] = (k < 64) ? Win[n][k] : W_res[n][k-64], bf16, row-major [R_][K_]
__global__ __launch_bounds__(256) void k_build_wcat(const float* __restrict__ Win,
                                                    const float* __restrict__ Wres,
                                                    __bf16* __restrict__ wcat) {
  size_t i = (size_t)blockIdx.x * 256 + threadIdx.x;
  if (i >= (size_t)R_ * K_) return;
  int n = (int)(i / K_);
  int k = (int)(i % K_);
  float v = (k < D_) ? Win[(size_t)n * D_ + k] : Wres[(size_t)n * R_ + (k - D_)];
  wcat[i] = (__bf16)v;
}

__global__ __launch_bounds__(256) void k_cvt_input(const float* __restrict__ x,
                                                   __bf16* __restrict__ xbf, int n) {
  int i = blockIdx.x * 256 + threadIdx.x;
  if (i < n) xbf[i] = (__bf16)x[i];
}

__global__ __launch_bounds__(256) void k_zero_r(float* __restrict__ rf,
                                                __bf16* __restrict__ rb) {
  int i = blockIdx.x * 256 + threadIdx.x;   // grid is exactly B_*R_/256
  rf[i] = 0.0f;
  rb[i] = (__bf16)0.0f;
}

// ---------------- recurrent step ----------------
// pre[b][n] = sum_k A[b][k] * Wcat[n][k],  A = [ x_bf16[b][t][:] | r_bf16[b][:] ]
// r_new = (1-lr)*r_old + lr*tanh(pre + bias)
// Grid: (4, 32) workgroups of 256 threads; WG tile = 64(M) x 64(N).
// 8 waves arranged 2(M)x4(N); each wave owns a 32x16 output => 2 WMMA accums.
__global__ __launch_bounds__(256, 2) void k_esn_step(
    const __bf16* __restrict__ xbf,    // [B_][T_][D_]
    const __bf16* __restrict__ wcat,   // [R_][K_]
    const float*  __restrict__ bias,   // [R_]
    const float*  __restrict__ rf_in,  // [B_][R_]
    const __bf16* __restrict__ rb_in,  // [B_][R_]
    float*        __restrict__ rf_out,
    __bf16*       __restrict__ rb_out,
    int t) {
  __shared__ __align__(16) __bf16 As[2][64 * 64];   // [buf][row][k-chunk 64] = 16 KB

  const int tid  = threadIdx.x;
  const int lane = tid & 31;
  const int wave = tid >> 5;
  const int wm   = wave & 1;    // which 32-row half of the WG tile
  const int wn   = wave >> 1;   // which 16-col quarter
  const int l15  = lane & 15;
  const int half = lane >> 4;

  const int m_base = blockIdx.x * 64;   // batch rows
  const int n_base = blockIdx.y * 64;   // reservoir cols

  // A staging: 256 threads cover 64 rows x 64 k, two b128 copies each
  const int arow = tid >> 2;            // 0..63
  const int acol = (tid & 3) * 8;       // 0,8,16,24
  const int gb   = m_base + arow;

  v8f acc0 = {};
  v8f acc1 = {};

  auto stageA = [&](int kc, int buf) {
#pragma unroll
    for (int seg = 0; seg < 2; ++seg) {
      const int kg = kc + acol + seg * 32;   // global K index
      const __bf16* src = (kg < D_)
          ? (xbf + (((size_t)gb * T_ + t) * D_ + kg))
          : (rb_in + ((size_t)gb * R_ + (kg - D_)));
      copy16_g2l(&As[buf][arow * 64 + acol + seg * 32], src);
    }
  };

  stageA(0, 0);

  // This wave/lane's B column, offset by the half-selected k group
  const __bf16* bcol =
      wcat + (size_t)(n_base + wn * 16 + l15) * K_ + half * 8;

  for (int stage = 0; stage < 33; ++stage) {     // 33 * 64 = 2112 = K_
    const int buf = stage & 1;
    wait_async_copies();
    __syncthreads();   // buf fully staged by all waves; prev buf fully consumed
    if (stage + 1 < 33) stageA(64 * (stage + 1), buf ^ 1);

    const int kc = stage * 64;
#pragma unroll
    for (int kk = 0; kk < 64; kk += 32) {
      FragAB a0, a1, b;
      const __bf16* ar0 = &As[buf][(wm * 32 + l15) * 64 + kk + half * 8];
      const __bf16* ar1 = &As[buf][(wm * 32 + 16 + l15) * 64 + kk + half * 8];
      a0.q[0] = *(const u32x4*)(ar0);        // k = kk + half*8 + 0..7
      a0.q[1] = *(const u32x4*)(ar0 + 16);   // k = kk + 16 + half*8 + 0..7
      a1.q[0] = *(const u32x4*)(ar1);
      a1.q[1] = *(const u32x4*)(ar1 + 16);
      const u32x4* bp = (const u32x4*)(bcol + kc + kk);  // k = half*8 + 0..7, 16 + half*8 + 0..7
      b.q[0] = bp[0];
      b.q[1] = bp[2];
      acc0 = __builtin_amdgcn_wmma_f32_16x16x32_bf16(
          false, a0.v, false, b.v, (short)0, acc0, false, false);
      acc1 = __builtin_amdgcn_wmma_f32_16x16x32_bf16(
          false, a1.v, false, b.v, (short)0, acc1, false, false);
    }
    if (stage + 1 < 33)
      __builtin_prefetch(bcol + (stage + 1) * 64, 0, 1);  // global_prefetch next B
  }

  // Fused epilogue: bias + tanh + leaky blend; write f32 state + bf16 shadow.
  const int   n_g = n_base + wn * 16 + l15;
  const float bn  = bias[n_g];
#pragma unroll
  for (int s = 0; s < 2; ++s) {
    const v8f acc = s ? acc1 : acc0;
    const int m0  = m_base + wm * 32 + s * 16 + half * 8;  // C layout: M = 8*half + i
#pragma unroll
    for (int i = 0; i < 8; ++i) {
      const size_t idx  = (size_t)(m0 + i) * R_ + n_g;
      const float  pre  = acc[i] + bn;
      const float  rold = rf_in[idx];
      const float  rnew = (1.0f - LRATE) * rold + LRATE * tanhf(pre);
      rf_out[idx] = rnew;
      rb_out[idx] = (__bf16)rnew;
    }
  }
}

// ---------------- readout ----------------
// One block per batch row: copies r_final into the output tail and computes
// the 10-wide readout via LDS tree reduction.
__global__ __launch_bounds__(256) void k_esn_output(const float* __restrict__ rf,
                                                    const float* __restrict__ Wout,
                                                    const float* __restrict__ bout,
                                                    float* __restrict__ out) {
  __shared__ float red[256];
  const int b   = blockIdx.x;
  const int tid = threadIdx.x;
  const float* r = rf + (size_t)b * R_;

  float* rtail = out + (size_t)B_ * O_ + (size_t)b * R_;
  for (int i = tid; i < R_; i += 256) rtail[i] = r[i];

  for (int o = 0; o < O_; ++o) {
    float p = 0.0f;
    for (int i = tid; i < R_; i += 256) p += Wout[(size_t)o * R_ + i] * r[i];
    red[tid] = p;
    __syncthreads();
    for (int s = 128; s > 0; s >>= 1) {
      if (tid < s) red[tid] += red[tid + s];
      __syncthreads();
    }
    if (tid == 0) out[b * O_ + o] = red[0] + bout[o];
    __syncthreads();
  }
}

// ---------------- host launcher ----------------
extern "C" void kernel_launch(void* const* d_in, const int* in_sizes, int n_in,
                              void* d_out, int out_size, void* d_ws, size_t ws_size,
                              hipStream_t stream) {
  (void)in_sizes; (void)n_in; (void)out_size; (void)ws_size;
  const float* x    = (const float*)d_in[0];  // [B_][T_][D_]
  const float* Win  = (const float*)d_in[1];  // [R_][D_]
  const float* Wres = (const float*)d_in[2];  // [R_][R_]
  const float* bres = (const float*)d_in[3];  // [R_]
  const float* Wout = (const float*)d_in[4];  // [O_][R_]
  const float* bout = (const float*)d_in[5];  // [O_]

  unsigned char* ws = (unsigned char*)d_ws;
  size_t off = 0;
  auto wsalloc = [&](size_t bytes) -> void* {
    void* p = ws + off;
    off = (off + bytes + 255) & ~(size_t)255;
    return p;
  };
  __bf16* wcat = (__bf16*)wsalloc((size_t)R_ * K_ * sizeof(__bf16));       // 8.25 MB
  __bf16* xbf  = (__bf16*)wsalloc((size_t)B_ * T_ * D_ * sizeof(__bf16));  // 16 MB
  float*  rf0  = (float*) wsalloc((size_t)B_ * R_ * sizeof(float));        // 2 MB
  float*  rf1  = (float*) wsalloc((size_t)B_ * R_ * sizeof(float));        // 2 MB
  __bf16* rb0  = (__bf16*)wsalloc((size_t)B_ * R_ * sizeof(__bf16));       // 1 MB
  __bf16* rb1  = (__bf16*)wsalloc((size_t)B_ * R_ * sizeof(__bf16));       // 1 MB

  {
    const size_t n = (size_t)R_ * K_;
    k_build_wcat<<<(unsigned)((n + 255) / 256), 256, 0, stream>>>(Win, Wres, wcat);
  }
  {
    const int n = B_ * T_ * D_;
    k_cvt_input<<<(n + 255) / 256, 256, 0, stream>>>(x, xbf, n);
  }
  k_zero_r<<<(B_ * R_) / 256, 256, 0, stream>>>(rf0, rb0);

  float*  rf[2] = {rf0, rf1};
  __bf16* rb[2] = {rb0, rb1};
  const dim3 grid(B_ / 64, R_ / 64);   // (4, 32)
  for (int t = 0; t < T_; ++t) {
    const int p = t & 1;
    k_esn_step<<<grid, 256, 0, stream>>>(xbf, wcat, bres,
                                         rf[p], rb[p], rf[p ^ 1], rb[p ^ 1], t);
  }
  // T_ = 512 is even -> final state lands back in rf[0]
  k_esn_output<<<B_, 256, 0, stream>>>(rf[0], Wout, bout, (float*)d_out);
}
